// Comatch_88880053223599
// MI455X (gfx1250) — compile-verified, matrix operand order checked by reading, not run on hardware
//
#include <hip/hip_runtime.h>

// Problem constants
#define NQ_  32
#define M_   32
#define ND_  200
#define NN_  160
#define H_   768
#define DIM_ 128
#define NEG_ (-10000.0f)

typedef __attribute__((ext_vector_type(16))) _Float16 v16h;
typedef __attribute__((ext_vector_type(8)))  float    v8f;

// Load a 16x32 f16 WMMA fragment from a row-major matrix (stride in halfs).
// ISA 16-bit A layout: lanes 0-15 hold row M=lane, halves {K0..7, K16..23};
// lanes 16-31 hold row M=lane-16, halves {K8..15, K24..31}. The dense B
// layout is the symmetric transpose (lane = N column), so the same loader
// serves B when the source matrix is stored as [N][K] row-major.
__device__ __forceinline__ v16h load_frag_h(const _Float16* base, int stride,
                                            int r0, int k0, int lane) {
  int row = r0 + (lane & 15);
  int ko  = k0 + ((lane & 16) ? 8 : 0);
  const _Float16* p = base + (size_t)row * stride + ko;
  union { uint4 u[2]; v16h v; } t;
  t.u[0] = *(const uint4*)(p);
  t.u[1] = *(const uint4*)(p + 16);
  return t.v;
}

// Same fragment, but sourced from f32 data with inline f32->f16 conversion
// (avoids a separate 98 MB conversion pass over Q/D).
__device__ __forceinline__ v16h load_frag_f32(const float* base, int stride,
                                              int r0, int k0, int lane) {
  int row = r0 + (lane & 15);
  int ko  = k0 + ((lane & 16) ? 8 : 0);
  const float* p = base + (size_t)row * stride + ko;
  float4 a0 = *(const float4*)(p);
  float4 a1 = *(const float4*)(p + 4);
  float4 b0 = *(const float4*)(p + 16);
  float4 b1 = *(const float4*)(p + 20);
  v16h v;
  v[0]=(_Float16)a0.x;  v[1]=(_Float16)a0.y;  v[2]=(_Float16)a0.z;  v[3]=(_Float16)a0.w;
  v[4]=(_Float16)a1.x;  v[5]=(_Float16)a1.y;  v[6]=(_Float16)a1.z;  v[7]=(_Float16)a1.w;
  v[8]=(_Float16)b0.x;  v[9]=(_Float16)b0.y;  v[10]=(_Float16)b0.z; v[11]=(_Float16)b0.w;
  v[12]=(_Float16)b1.x; v[13]=(_Float16)b1.y; v[14]=(_Float16)b1.z; v[15]=(_Float16)b1.w;
  return v;
}

// Kernel 0: W (H x DIM, f32) -> Wt (DIM x H, f16) so projection B-fragments
// are contiguous 16-byte loads.
__global__ void comatch_wt_kernel(const float* __restrict__ W,
                                  _Float16* __restrict__ Wt) {
  int idx = blockIdx.x * blockDim.x + threadIdx.x;
  if (idx < H_ * DIM_) {
    int h = idx / DIM_;
    int d = idx - h * DIM_;
    Wt[(size_t)d * H_ + h] = (_Float16)W[idx];
  }
}

// Kernel 1: project 16-row strips of Q or D through W (K=768) with WMMA,
// l2-normalize each 128-wide row, store f16. One wave per strip.
__global__ void __launch_bounds__(32)
comatch_proj_kernel(const float* __restrict__ Q, const float* __restrict__ D,
                    const _Float16* __restrict__ Wt,
                    _Float16* __restrict__ Qp, _Float16* __restrict__ Dp) {
  __shared__ float tile[16 * 129];   // stride 129 (odd) -> conflict-free
  __shared__ float rscale[16];
  const int lane = threadIdx.x;
  const int s = blockIdx.x;

  const float* src; int r0; _Float16* dst;
  if (s < (NQ_ * M_) / 16) { src = Q; r0 = s * 16;                    dst = Qp; }
  else                     { src = D; r0 = (s - (NQ_ * M_) / 16) * 16; dst = Dp; }

  const v8f zero8 = {0.f,0.f,0.f,0.f,0.f,0.f,0.f,0.f};
  v8f acc[8];
#pragma unroll
  for (int nt = 0; nt < 8; ++nt) acc[nt] = zero8;

  for (int k0 = 0; k0 < H_; k0 += 32) {
    v16h a = load_frag_f32(src, H_, r0, k0, lane);
#pragma unroll
    for (int nt = 0; nt < 8; ++nt) {
      v16h b = load_frag_h(Wt, H_, nt * 16, k0, lane);
      acc[nt] = __builtin_amdgcn_wmma_f32_16x16x32_f16(
          false, a, false, b, (short)0, acc[nt], false, false);
    }
  }

  // C layout: VGPR r, lanes 0-15 -> (M=r, N=lane); lanes 16-31 -> (M=r+8, N=lane-16)
#pragma unroll
  for (int nt = 0; nt < 8; ++nt)
#pragma unroll
    for (int r = 0; r < 8; ++r) {
      int m = (lane & 16) ? (r + 8) : r;
      tile[m * 129 + nt * 16 + (lane & 15)] = acc[nt][r];
    }
  __syncthreads();

  if (lane < 16) {
    float ss = 0.f;
    for (int c = 0; c < DIM_; ++c) { float v = tile[lane * 129 + c]; ss += v * v; }
    rscale[lane] = 1.0f / fmaxf(sqrtf(ss), 1e-12f);
  }
  __syncthreads();

  for (int i = 0; i < (16 * DIM_) / 32; ++i) {
    int idx = i * 32 + lane;
    int row = idx >> 7, col = idx & 127;
    dst[(size_t)(r0 + row) * DIM_ + col] =
        (_Float16)(tile[row * 129 + col] * rscale[row]);
  }
}

// Kernel 2: one wave per (q,d). The 20 KB gumbel tile is streamed to LDS with
// CDNA5 GLOBAL_LOAD_ASYNC_TO_LDS_B128 (ASYNCcnt-tracked, bypasses VGPRs) and
// overlaps the 80 WMMAs computing sim = Qp[q] x Dp[d]^T. Then per-m streaming
// argmax of sim/TEMP + gumbel (straight-through forward == one-hot at argmax)
// and a q_mask-weighted wave32 reduction.
__global__ void __launch_bounds__(32)
comatch_score_kernel(const _Float16* __restrict__ Qp, const _Float16* __restrict__ Dp,
                     const float* __restrict__ q_mask, const float* __restrict__ d_mask,
                     const float* __restrict__ gumbel, float* __restrict__ out) {
  __shared__ float sim[M_ * 161];  // stride 161 (odd) -> conflict-free rows
  __shared__ float gn[M_ * 161];
  const int lane = threadIdx.x;
  const int blk = blockIdx.x;
  const int q = blk / ND_;
  const int d = blk - q * ND_;

  // Kick off the async gumbel stream: 40 per-lane 16B chunks. Each chunk is 4
  // floats that never straddle a 160-float row (160 % 4 == 0), so it scatters
  // cleanly into the padded stride-161 LDS layout. Generic shared pointers put
  // the LDS aperture in addr[63:32], so the low 32 bits are the wave-relative
  // LDS offset the async VDST operand expects.
  const float* g = gumbel + (size_t)blk * (M_ * NN_);
#pragma unroll 4
  for (int i = 0; i < (M_ * NN_) / (32 * 4); ++i) {
    int f = (i * 32 + lane) * 4;             // starting float of this chunk
    int m = f / NN_;
    int n = f - m * NN_;
    const float* gp = g + f;
    unsigned lp = (unsigned)(size_t)(&gn[m * 161 + n]);
    asm volatile("global_load_async_to_lds_b128 %0, %1, off"
                 :: "v"(lp), "v"(gp) : "memory");
  }

  // sim tiles (async gumbel copies are in flight underneath these WMMAs)
#pragma unroll
  for (int mt = 0; mt < 2; ++mt) {
    v16h afr[4];
#pragma unroll
    for (int kk = 0; kk < 4; ++kk)
      afr[kk] = load_frag_h(Qp, DIM_, q * M_ + mt * 16, kk * 32, lane);
#pragma unroll
    for (int nt = 0; nt < 10; ++nt) {
      v8f acc = {0.f,0.f,0.f,0.f,0.f,0.f,0.f,0.f};
#pragma unroll
      for (int kk = 0; kk < 4; ++kk) {
        v16h b = load_frag_h(Dp, DIM_, d * NN_ + nt * 16, kk * 32, lane);
        acc = __builtin_amdgcn_wmma_f32_16x16x32_f16(
            false, afr[kk], false, b, (short)0, acc, false, false);
      }
      int n = nt * 16 + (lane & 15);
      float dm  = d_mask[d * NN_ + n];
      float neg = (1.0f - dm) * NEG_;           // exact reference masking
#pragma unroll
      for (int r = 0; r < 8; ++r) {
        int m = mt * 16 + ((lane & 16) ? (r + 8) : r);
        sim[m * 161 + n] = acc[r] * dm + neg;
      }
    }
  }

  // Drain the async gumbel stream, then sync the (single-wave) block.
  asm volatile("s_wait_asynccnt 0" ::: "memory");
  __syncthreads();

  // lane == m; strict '>' keeps first max index like jnp.argmax.
  const int m = lane;
  float best_key = -3.4e38f, best_val = 0.f;
  for (int n = 0; n < NN_; ++n) {
    float s   = sim[m * 161 + n];
    float key = s * 10.0f + gn[m * 161 + n];    // 1/TEMP = 10
    if (key > best_key) { best_key = key; best_val = s; }
  }
  float sc = q_mask[q * M_ + m] * best_val;
#pragma unroll
  for (int off = 16; off > 0; off >>= 1) sc += __shfl_xor(sc, off, 32);
  if (lane == 0) out[blk] = sc;
}

extern "C" void kernel_launch(void* const* d_in, const int* in_sizes, int n_in,
                              void* d_out, int out_size, void* d_ws, size_t ws_size,
                              hipStream_t stream) {
  const float* Q      = (const float*)d_in[0];  // (32,32,768)
  const float* D      = (const float*)d_in[1];  // (200,160,768)
  const float* q_mask = (const float*)d_in[2];  // (32,32)
  const float* d_mask = (const float*)d_in[3];  // (200,160)
  const float* gumbel = (const float*)d_in[4];  // (32,200,32,160)
  const float* W      = (const float*)d_in[5];  // (768,128)
  float* out = (float*)d_out;                   // (32,200)

  // Workspace partition (all offsets 16B-aligned):
  //   Wt: 128*768*2 = 196608 B, Qp: 1024*128*2 = 262144 B, Dp: 32000*128*2 = 8192000 B
  char* ws = (char*)d_ws;
  _Float16* Wt = (_Float16*)(ws);
  _Float16* Qp = (_Float16*)(ws + 196608);
  _Float16* Dp = (_Float16*)(ws + 196608 + 262144);

  comatch_wt_kernel<<<(H_ * DIM_ + 255) / 256, 256, 0, stream>>>(W, Wt);
  comatch_proj_kernel<<<64 + 2000, 32, 0, stream>>>(Q, D, Wt, Qp, Dp);
  comatch_score_kernel<<<NQ_ * ND_, 32, 0, stream>>>(Qp, Dp, q_mask, d_mask, gumbel, out);
}
